// SparseConv2d_89421219103271
// MI455X (gfx1250) — matrix operand order repaired and checked by reading, not compiled, not run on hardware
//
#include <hip/hip_runtime.h>
#include <hip/hip_bf16.h>

typedef _Float16 v16h __attribute__((ext_vector_type(16)));
typedef _Float16 v8h  __attribute__((ext_vector_type(8)));
typedef float    v8f  __attribute__((ext_vector_type(8)));

#define CIN   128
#define COUT  128
#define HH    128
#define WW    128
#define NB    16
#define SPT   32          // sparsity tile edge
#define THRESH 1e-6f

// LDS x staging: 4 rows x 72 cols x 128 ci (f16), col stride padded to 136 halves
// (272B, multiple of 16B so ds_load_b128 stays aligned).
// Staged col cc corresponds to image column wc = wbase - 4 + cc  (4-aligned start
// so all global float4 loads are 16B aligned and never straddle W or a 32-tile).
#define COLS   72
#define CSTR   136
#define ROWH   (COLS * CSTR)               // 9792 halves per staged row
#define LDS_BYTES (4 * ROWH * 2)           // 78336 bytes (dynamic LDS)
#define WFRAG_BYTES (9 * 4 * 8 * 32 * 16 * 2)   // 294912

// ---------------------------------------------------------------------------
// Kernel 1: swizzle fp32 weights into WMMA A-fragment lane layout (f16).
// Layout: [khw 0..8][cib 0..3][co_blk 0..7][lane 0..31][j 0..15]
// A (16x32 f16) per ISA: lane L -> M = L%16; halves j: K = (j<8 ? j : j+8) + (L<16?0:8)
// ---------------------------------------------------------------------------
__global__ void wprep_kernel(const float* __restrict__ w, _Float16* __restrict__ wf) {
    int e = blockIdx.x * 256 + threadIdx.x;
    if (e >= 9 * 4 * 8 * 32 * 16) return;
    int j    = e & 15;
    int lane = (e >> 4) & 31;
    int rest = e >> 9;
    int co_blk = rest & 7;
    int cib    = (rest >> 3) & 3;
    int khw    = rest >> 5;                 // 0..8
    int hi = lane >> 4;
    int M  = lane & 15;
    int K  = (j < 8 ? j : j + 8) + hi * 8;
    int co = co_blk * 16 + M;
    int ci = cib * 32 + K;
    int kh = khw / 3, kw = khw % 3;
    wf[e] = (_Float16)w[((co * CIN + ci) * 3 + kh) * 3 + kw];
}

// ---------------------------------------------------------------------------
// Kernel 2: 32x32 spatial tile activity mask (any |x| > THRESH across 128 ch).
// One block per (n, th, tw); early-exit scan.
// ---------------------------------------------------------------------------
__global__ void mask_kernel(const float* __restrict__ x, int* __restrict__ mask) {
    int b  = blockIdx.x;
    int n  = b >> 4;
    int th = (b >> 2) & 3;
    int tw = b & 3;
    __shared__ int s;
    if (threadIdx.x == 0) s = 0;
    __syncthreads();
    const float* xb = x + (size_t)n * CIN * HH * WW + th * SPT * WW + tw * SPT;
    int found = 0;
    for (int ci = 0; ci < CIN && !found; ++ci) {
        for (int p = threadIdx.x; p < SPT * SPT; p += 256) {
            int ph = p >> 5, pw = p & 31;
            float v = xb[(size_t)ci * (HH * WW) + ph * WW + pw];
            if (fabsf(v) > THRESH) { found = 1; break; }
        }
    }
    if (found) atomicOr(&s, 1);
    __syncthreads();
    if (threadIdx.x == 0) mask[b] = s;
}

// ---------------------------------------------------------------------------
// Kernel 3: implicit-GEMM conv via V_WMMA_F32_16X16X32_F16.
// Block = 256 thr (8 waves). WG tile: 128 co x 64 px x 2 output rows.
// Wave w -> co block [16w,16w+16); 8 accumulators (2 rows x 4 pixel tiles).
// ---------------------------------------------------------------------------
__global__ __launch_bounds__(256)
void conv_wmma_kernel(const float* __restrict__ x,
                      const _Float16* __restrict__ wf,
                      const int* __restrict__ mask,
                      const float* __restrict__ bias,
                      float* __restrict__ out) {
    extern __shared__ char smem_raw[];
    _Float16* sx = (_Float16*)smem_raw;

    const int tid   = threadIdx.x;
    const int n     = blockIdx.z;
    const int h0    = blockIdx.y * 2;
    const int wbase = blockIdx.x * 64;

    // ---- uniform sparse fast path: all touched 32x32 tiles inactive -> bias ----
    {
        int th0 = (h0 > 0 ? h0 - 1 : 0) >> 5;
        int th1 = (h0 + 2 < HH ? h0 + 2 : HH - 1) >> 5;
        int tw0 = (wbase > 0 ? wbase - 1 : 0) >> 5;
        int tw1 = (wbase + 64 < WW ? wbase + 64 : WW - 1) >> 5;
        int any = 0;
        for (int th = th0; th <= th1; ++th)
            for (int tw = tw0; tw <= tw1; ++tw)
                any |= mask[(n * 4 + th) * 4 + tw];
        if (!any) {     // uniform across workgroup: no divergence around barriers
            for (int idx = tid; idx < 2 * COUT * 64; idx += 256) {
                int r0  = idx >> 13;
                int rem = idx & 8191;
                int co  = rem >> 6;
                int px  = rem & 63;
                out[(((size_t)n * COUT + co) * HH + h0 + r0) * WW + wbase + px] = bias[co];
            }
            return;
        }
    }

    const float* xn = x + (size_t)n * CIN * HH * WW;

    // ---- stage x rows h0-1..h0+2 into LDS as f16, [col][ci], float4 quads ----
    for (int r = 0; r < 4; ++r) {
        int hr = h0 + r - 1;
        int rowok = (hr >= 0 && hr < HH);
        int th = (hr & 127) >> 5;           // valid only when rowok
        for (int q = tid; q < 18 * CIN; q += 256) {
            int ci  = q / 18;
            int qq  = q - ci * 18;
            int wc0 = wbase - 4 + qq * 4;   // 4-aligned; fully in or fully out
            float4 v = make_float4(0.f, 0.f, 0.f, 0.f);
            if (rowok && wc0 >= 0 && wc0 < WW) {
                int tw = wc0 >> 5;
                if (mask[(n * 4 + th) * 4 + tw])
                    v = *(const float4*)(xn + (size_t)ci * (HH * WW) + hr * WW + wc0);
            }
            int base = r * ROWH + (qq * 4) * CSTR + ci;
            sx[base           ] = (_Float16)v.x;
            sx[base +     CSTR] = (_Float16)v.y;
            sx[base + 2 * CSTR] = (_Float16)v.z;
            sx[base + 3 * CSTR] = (_Float16)v.w;
        }
    }
    __syncthreads();

    // ---- per-wave WMMA GEMM ----
    const int wv   = tid >> 5;     // co block 0..7
    const int lane = tid & 31;
    const int lo   = lane & 15;    // N (pixel) / M index
    const int hi   = lane >> 4;

    v8f acc[8];
#pragma unroll
    for (int t = 0; t < 8; ++t) acc[t] = (v8f){0.f,0.f,0.f,0.f,0.f,0.f,0.f,0.f};

    const _Float16* wfl = wf + ((size_t)wv * 32 + lane) * 16;

    for (int kh = 0; kh < 3; ++kh) {
        for (int cib = 0; cib < 4; ++cib) {
#pragma unroll
            for (int kw = 0; kw < 3; ++kw) {
                const int khw = kh * 3 + kw;
                const _Float16* ap = wfl + (size_t)(khw * 4 + cib) * (8 * 32 * 16);
                v16h a = *(const v16h*)ap;
                __builtin_prefetch(ap + 8 * 32 * 16, 0, 3);   // next cib fragment (L2)
#pragma unroll
                for (int r0 = 0; r0 < 2; ++r0) {
                    const int bbase = (r0 + kh) * ROWH + cib * 32 + hi * 8;
#pragma unroll
                    for (int t = 0; t < 4; ++t) {
                        const int col = t * 16 + lo + kw + 3;    // 3..68 < 72
                        const _Float16* bp = &sx[bbase + col * CSTR];
                        v8h blo = *(const v8h*)bp;
                        v8h bhi = *(const v8h*)(bp + 16);
                        v16h b;
#pragma unroll
                        for (int i = 0; i < 8; ++i) { b[i] = blo[i]; b[i + 8] = bhi[i]; }
                        acc[r0 * 4 + t] = __builtin_amdgcn_wmma_f32_16x16x32_f16(
                            false, a, false, b, (short)0, acc[r0 * 4 + t], false, false);
                    }
                }
            }
        }
    }

    // ---- bias + store: lane elem v -> co = wv*16 + hi*8 + v, pixel = lo ----
    const int co0 = wv * 16 + hi * 8;
    __attribute__((aligned(16))) float bb[8];
    const float4* b4 = (const float4*)(bias + co0);
    *(float4*)&bb[0] = b4[0];
    *(float4*)&bb[4] = b4[1];

#pragma unroll
    for (int r0 = 0; r0 < 2; ++r0) {
#pragma unroll
        for (int t = 0; t < 4; ++t) {
            const int wc = wbase + t * 16 + lo;
#pragma unroll
            for (int v = 0; v < 8; ++v) {
                out[(((size_t)n * COUT + (co0 + v)) * HH + h0 + r0) * WW + wc] =
                    acc[r0 * 4 + t][v] + bb[v];
            }
        }
    }
}

// ---------------------------------------------------------------------------
extern "C" void kernel_launch(void* const* d_in, const int* in_sizes, int n_in,
                              void* d_out, int out_size, void* d_ws, size_t ws_size,
                              hipStream_t stream) {
    const float* x    = (const float*)d_in[0];
    const float* w    = (const float*)d_in[1];
    const float* bias = (const float*)d_in[2];
    float* out        = (float*)d_out;

    _Float16* wfrag = (_Float16*)d_ws;
    int* mask       = (int*)((char*)d_ws + WFRAG_BYTES);

    // 1) swizzle weights into WMMA A-fragment layout (147456 halves)
    wprep_kernel<<<(9 * 4 * 8 * 32 * 16 + 255) / 256, 256, 0, stream>>>(w, wfrag);
    // 2) activity mask per 32x32 tile (16*4*4 blocks)
    mask_kernel<<<NB * 4 * 4, 256, 0, stream>>>(x, mask);
    // 3) WMMA conv: grid (W/64, H/2, N), dynamic LDS 78336B
    conv_wmma_kernel<<<dim3(WW / 64, HH / 2, NB), 256, LDS_BYTES, stream>>>(
        x, wfrag, mask, bias, out);
}
